// CausalSelfAttention_86775519248666
// MI455X (gfx1250) — matrix operand order repaired and checked by reading, not compiled, not run on hardware
//
#include <hip/hip_runtime.h>
#include <hip/hip_bf16.h>

// ---------------------------------------------------------------------------
// CausalSelfAttention for MI455X (gfx1250, wave32, WMMA bf16 path)
// B=4, T=2048, C=1024, H=16, hd=64
// ---------------------------------------------------------------------------

typedef __bf16 bf16_t;
typedef __attribute__((ext_vector_type(16))) __bf16 v16bf;
typedef __attribute__((ext_vector_type(8)))  __bf16 v8bf;
typedef __attribute__((ext_vector_type(8)))  float  v8f;
typedef __attribute__((ext_vector_type(4)))  int    v4i;

#define WMMA_BF16(a, b, c) \
  __builtin_amdgcn_wmma_f32_16x16x32_bf16(false, (a), false, (b), (short)0, (c), false, false)

// ---- CDNA5 async global->LDS copy (ASYNCcnt path), with safe fallback -----
#if __has_builtin(__builtin_amdgcn_global_load_async_to_lds_b128) && \
    __has_builtin(__builtin_amdgcn_s_wait_asynccnt)
#define HAVE_ASYNC_LDS 1
#else
#define HAVE_ASYNC_LDS 0
#endif

typedef __attribute__((address_space(1))) v4i g_v4i;   // global
typedef __attribute__((address_space(3))) v4i l_v4i;   // LDS

static __device__ __forceinline__ void copy16_to_lds(bf16_t* lds_dst,
                                                     const bf16_t* gsrc) {
#if HAVE_ASYNC_LDS
  __builtin_amdgcn_global_load_async_to_lds_b128(
      (g_v4i*)(uintptr_t)gsrc, (l_v4i*)(uintptr_t)lds_dst, 0, 0);
#else
  *(uint4*)lds_dst = *(const uint4*)gsrc;
#endif
}

static __device__ __forceinline__ void wait_async_lds() {
#if HAVE_ASYNC_LDS
  __builtin_amdgcn_s_wait_asynccnt(0);
#endif
}

// ---- WMMA fragment gathers from LDS (per ISA 7.12.2 layouts, wave32) ------
// Both halves of a fragment are 16 contiguous bytes -> two ds_load_b128.
static __device__ __forceinline__ v16bf frag_pair(const bf16_t* p, int gap) {
  v8bf lo = *(const v8bf*)p;
  v8bf hi = *(const v8bf*)(p + gap);
  return __builtin_shufflevector(lo, hi, 0, 1, 2, 3, 4, 5, 6, 7,
                                 8, 9, 10, 11, 12, 13, 14, 15);
}

// A 16x32 bf16: lane half h, r=lane%16 owns row r.
//   elements 0..7  -> K = 8h .. 8h+7        (16 contiguous bytes)
//   elements 8..15 -> K = 16+8h .. 16+8h+7  (16 contiguous bytes)
static __device__ __forceinline__ v16bf frag_a_bf16(const bf16_t* lds, int ld,
                                                    int row_base, int kofs, int lane) {
  const int h = (lane >> 4) & 1;
  const int r = lane & 15;
  const bf16_t* p = lds + (size_t)(row_base + r) * ld + kofs + 8 * h;
  return frag_pair(p, 16);
}

// B 32x16 bf16 from N-major LDS tile lds[n][k]: lane half h owns K
// 8h..8h+7 and 16+8h..16+8h+7 (two 16B chunks), column N = lane%16.
static __device__ __forceinline__ v16bf frag_b_bf16(const bf16_t* lds, int ld,
                                                    int n_base, int kofs, int lane) {
  const int h = (lane >> 4) & 1;
  const int r = lane & 15;
  const bf16_t* p = lds + (size_t)(n_base + r) * ld + kofs + 8 * h;
  return frag_pair(p, 16);
}

// ---------------------------------------------------------------------------
// f32 -> bf16 conversion (row-major copy)
// ---------------------------------------------------------------------------
__global__ void cvt_f32_to_bf16(const float* __restrict__ in,
                                bf16_t* __restrict__ out, int n) {
  int i = blockIdx.x * 256 + threadIdx.x;
  if (i < n) out[i] = (bf16_t)in[i];
}

// f32 [K][N] -> bf16 [N][K]  (one-off weight transpose so GEMM B-tiles are
// contiguous 16B rows and can be staged with async b128 copies)
__global__ void cvt_transpose_bf16(const float* __restrict__ in,
                                   bf16_t* __restrict__ out, int K, int N) {
  int idx = blockIdx.x * 256 + threadIdx.x;
  if (idx >= K * N) return;
  int k = idx / N, n = idx - k * N;          // reads coalesced along n
  out[(size_t)n * K + k] = (bf16_t)in[idx];
}

// ---------------------------------------------------------------------------
// Tiled GEMM: C[M,N] = A[M,K] (bf16 row-major) * Bt[N,K]^T (bf16) + bias
// Tile 128x128x32; 256 threads = 8 waves; wave computes 32x64 via 2x4 WMMA.
// Double-buffered LDS with async global->LDS prefetch of the next K-tile.
// ---------------------------------------------------------------------------
__global__ void gemm_bf16_wmma(const bf16_t* __restrict__ A,
                               const bf16_t* __restrict__ Bt,
                               const float* __restrict__ bias,
                               bf16_t* __restrict__ out_bf,
                               float* __restrict__ out_f32,
                               int M, int N, int K) {
  __shared__ __align__(16) bf16_t sA[2][128][40];  // [m][k], 80B rows (16B aligned)
  __shared__ __align__(16) bf16_t sB[2][128][40];  // [n][k]

  const int tid  = threadIdx.x;
  const int lane = tid & 31;
  const int wave = tid >> 5;       // 0..7
  const int wm   = wave & 3;       // row block (4 x 32 = 128)
  const int wn   = wave >> 2;      // col block (2 x 64 = 128)

  const int n0 = blockIdx.x * 128;
  const int m0 = blockIdx.y * 128;

  v8f acc[2][4];
#pragma unroll
  for (int i = 0; i < 2; ++i)
#pragma unroll
    for (int t = 0; t < 4; ++t) acc[i][t] = (v8f){};

  // Stage one K-tile (128 rows x 32 cols, 16B chunks; 512 chunks / operand)
  auto stage = [&](int buf, int k0) {
#pragma unroll
    for (int c = tid; c < 512; c += 256) {
      int row = c >> 2, q = (c & 3) * 8;
      copy16_to_lds(&sA[buf][row][q], A + (size_t)(m0 + row) * K + k0 + q);
      copy16_to_lds(&sB[buf][row][q], Bt + (size_t)(n0 + row) * K + k0 + q);
    }
  };

  stage(0, 0);
  wait_async_lds();
  __syncthreads();

  int buf = 0;
  for (int k0 = 0; k0 < K; k0 += 32, buf ^= 1) {
    if (k0 + 32 < K) stage(buf ^ 1, k0 + 32);   // async prefetch next tile

    v16bf a0 = frag_a_bf16(&sA[buf][0][0], 40, wm * 32, 0, lane);
    v16bf a1 = frag_a_bf16(&sA[buf][0][0], 40, wm * 32 + 16, 0, lane);
#pragma unroll
    for (int t = 0; t < 4; ++t) {
      v16bf b = frag_b_bf16(&sB[buf][0][0], 40, wn * 64 + 16 * t, 0, lane);
      acc[0][t] = WMMA_BF16(a0, b, acc[0][t]);
      acc[1][t] = WMMA_BF16(a1, b, acc[1][t]);
    }

    wait_async_lds();
    __syncthreads();
  }

  // Epilogue: C/D layout -> element e is row e + 8*half, col lane%16
  const int h = (lane >> 4) & 1;
  const int r = lane & 15;
#pragma unroll
  for (int i = 0; i < 2; ++i) {
#pragma unroll
    for (int t = 0; t < 4; ++t) {
#pragma unroll
      for (int e = 0; e < 8; ++e) {
        int row = m0 + wm * 32 + i * 16 + e + 8 * h;
        int col = n0 + wn * 64 + t * 16 + r;
        float v = acc[i][t][e] + bias[col];
        if (out_bf) out_bf[(size_t)row * N + col] = (bf16_t)v;
        else        out_f32[(size_t)row * N + col] = v;
      }
    }
  }
}

// ---------------------------------------------------------------------------
// Flash-style causal attention.
// qkv: bf16 [B*T, 3C] row-major.  y: bf16 [B*T, C] (head-concat layout).
// One block = 128 threads (4 waves) handles one (b, h, 64-row q tile).
// Wave w owns q rows w*16 .. w*16+15.
// ---------------------------------------------------------------------------
__global__ void attention_wmma(const bf16_t* __restrict__ qkv,
                               bf16_t* __restrict__ y) {
  constexpr int T = 2048, C3 = 3072, C = 1024, HD = 64;
  __shared__ __align__(16) bf16_t sQ[64][72];    // [q][d]
  __shared__ __align__(16) bf16_t sK[64][72];    // [key][d]  (n-major for QK^T)
  __shared__ __align__(16) bf16_t sVT[64][72];   // [d][key]  (n-major for PV)
  __shared__ __align__(16) bf16_t sP[64][72];    // probabilities (bf16)
  __shared__ float  sS[64][66];                  // raw scores (f32)
  __shared__ float  sCorr[64];                   // correction factor / inv-sum

  const int tid  = threadIdx.x;
  const int lane = tid & 31;
  const int w    = tid >> 5;       // wave 0..3
  const int h16  = (lane >> 4) & 1;
  const int r    = lane & 15;

  const int qt = blockIdx.x;       // q tile (64 rows)
  const int hh = blockIdx.y;       // head
  const int b  = blockIdx.z;       // batch

  // Stage Q tile once (rows are 128B contiguous -> async b128 chunks)
  for (int c = tid; c < 512; c += 128) {
    int row = c >> 3, q = (c & 7) * 8;
    copy16_to_lds(&sQ[row][q],
                  qkv + (size_t)(b * T + qt * 64 + row) * C3 + hh * HD + q);
  }
  wait_async_lds();
  __syncthreads();

  v16bf aq0 = frag_a_bf16(&sQ[0][0], 72, w * 16, 0, lane);
  v16bf aq1 = frag_a_bf16(&sQ[0][0], 72, w * 16, 32, lane);

  v8f yacc[4];
#pragma unroll
  for (int t = 0; t < 4; ++t) yacc[t] = (v8f){};

  float m_row = -3.0e38f;   // thread tid (<64) owns row tid
  float l_row = 0.0f;

  for (int kt = 0; kt <= qt; ++kt) {
    __syncthreads();  // protect LDS tiles from previous iteration's readers

    // K tile: async b128; V tile: manual transposed store [d][key]
    for (int c = tid; c < 512; c += 128) {
      int row = c >> 3, q = (c & 7) * 8;
      copy16_to_lds(&sK[row][q],
                    qkv + (size_t)(b * T + kt * 64 + row) * C3 + C + hh * HD + q);
    }
    for (int i = tid; i < 64 * 64; i += 128) {
      int kl = i >> 6, d = i & 63;
      sVT[d][kl] =
          qkv[(size_t)(b * T + kt * 64 + kl) * C3 + 2 * C + hh * HD + d];
    }
    wait_async_lds();
    __syncthreads();

    // S = Q K^T  (per wave: 16 q rows x 64 keys, K-dim = hd = 64)
    v8f sacc[4];
#pragma unroll
    for (int t = 0; t < 4; ++t) sacc[t] = (v8f){};
#pragma unroll
    for (int t = 0; t < 4; ++t) {
      v16bf bk0 = frag_b_bf16(&sK[0][0], 72, 16 * t, 0, lane);
      v16bf bk1 = frag_b_bf16(&sK[0][0], 72, 16 * t, 32, lane);
      sacc[t] = WMMA_BF16(aq0, bk0, sacc[t]);
      sacc[t] = WMMA_BF16(aq1, bk1, sacc[t]);
    }

    // Scale + causal mask, spill to LDS (f32)
#pragma unroll
    for (int t = 0; t < 4; ++t) {
#pragma unroll
      for (int e = 0; e < 8; ++e) {
        int row = w * 16 + e + 8 * h16;       // 0..63
        int qg  = qt * 64 + row;
        int kg  = kt * 64 + 16 * t + r;
        float v = sacc[t][e] * 0.125f;        // 1/sqrt(64)
        if (kg > qg) v = -3.0e38f;
        sS[row][16 * t + r] = v;
      }
    }
    __syncthreads();

    // Online softmax: thread tid (<64) owns row tid; write P as bf16
    if (tid < 64) {
      float mx = m_row;
#pragma unroll 4
      for (int c = 0; c < 64; ++c) mx = fmaxf(mx, sS[tid][c]);
      float corr = __expf(m_row - mx);
      float s = 0.0f;
#pragma unroll 4
      for (int c = 0; c < 64; ++c) {
        float e = __expf(sS[tid][c] - mx);
        sP[tid][c] = (bf16_t)e;
        s += e;
      }
      l_row = l_row * corr + s;
      m_row = mx;
      sCorr[tid] = corr;
    }
    __syncthreads();

    // Rescale running output by per-row correction factor
#pragma unroll
    for (int e = 0; e < 8; ++e) {
      float f = sCorr[w * 16 + e + 8 * h16];
#pragma unroll
      for (int t = 0; t < 4; ++t) yacc[t][e] *= f;
    }

    // Y += P V   (K-dim = 64 keys, two 32-wide steps)
#pragma unroll
    for (int ks = 0; ks < 2; ++ks) {
      v16bf ap = frag_a_bf16(&sP[0][0], 72, w * 16, 32 * ks, lane);
#pragma unroll
      for (int t = 0; t < 4; ++t) {
        v16bf bv = frag_b_bf16(&sVT[0][0], 72, 16 * t, 32 * ks, lane);
        yacc[t] = WMMA_BF16(ap, bv, yacc[t]);
      }
    }
  }

  __syncthreads();
  if (tid < 64) sCorr[tid] = 1.0f / l_row;   // reuse as inverse row sum
  __syncthreads();

  // Write y (bf16), layout [b, t, h*64 + d]
#pragma unroll
  for (int e = 0; e < 8; ++e) {
    int ql = w * 16 + e + 8 * h16;
    float inv = sCorr[ql];
#pragma unroll
    for (int t = 0; t < 4; ++t) {
      int d = 16 * t + r;
      float v = yacc[t][e] * inv;
      y[(size_t)(b * T + qt * 64 + ql) * C + hh * HD + d] = (bf16_t)v;
    }
  }
}

// ---------------------------------------------------------------------------
// Host-side launch
// ---------------------------------------------------------------------------
extern "C" void kernel_launch(void* const* d_in, const int* in_sizes, int n_in,
                              void* d_out, int out_size, void* d_ws, size_t ws_size,
                              hipStream_t stream) {
  constexpr int B = 4, T = 2048, C = 1024;
  constexpr int M = B * T;        // 8192
  constexpr int N_QKV = 3 * C;    // 3072

  const float* x      = (const float*)d_in[0];
  const float* W_attn = (const float*)d_in[1];
  const float* b_attn = (const float*)d_in[2];
  const float* W_proj = (const float*)d_in[3];
  const float* b_proj = (const float*)d_in[4];
  float* out = (float*)d_out;

  // Workspace layout (bytes)
  char* ws = (char*)d_ws;
  bf16_t* xb   = (bf16_t*)(ws);                      // 8192*1024*2  = 16,777,216
  bf16_t* wAt  = (bf16_t*)(ws + 16777216);           // [3072][1024] =  6,291,456
  bf16_t* wPt  = (bf16_t*)(ws + 23068672);           // [1024][1024] =  2,097,152
  bf16_t* qkvb = (bf16_t*)(ws + 25165824);           // 8192*3072*2  = 50,331,648
  bf16_t* yb   = (bf16_t*)(ws + 75497472);           // 8192*1024*2  = 16,777,216

  // 1) Convert x to bf16; transpose weights to [N][K] bf16
  {
    int n = M * C;
    cvt_f32_to_bf16<<<(n + 255) / 256, 256, 0, stream>>>(x, xb, n);
    n = C * N_QKV;
    cvt_transpose_bf16<<<(n + 255) / 256, 256, 0, stream>>>(W_attn, wAt, C, N_QKV);
    n = C * C;
    cvt_transpose_bf16<<<(n + 255) / 256, 256, 0, stream>>>(W_proj, wPt, C, C);
  }

  // 2) QKV projection: qkv = x @ W_attn + b_attn   (bf16 out)
  {
    dim3 grid(N_QKV / 128, M / 128);   // (24, 64)
    gemm_bf16_wmma<<<grid, 256, 0, stream>>>(xb, wAt, b_attn, qkvb, nullptr,
                                             M, N_QKV, C);
  }

  // 3) Causal attention
  {
    dim3 grid(T / 64, 16, B);          // (32, 16, 4)
    attention_wmma<<<grid, 128, 0, stream>>>(qkvb, yb);
  }

  // 4) Output projection: out = y @ W_proj + b_proj   (f32 out)
  {
    dim3 grid(C / 128, M / 128);       // (8, 64)
    gemm_bf16_wmma<<<grid, 256, 0, stream>>>(yb, wPt, b_proj, nullptr, out,
                                             M, C, C);
  }
}
